// AttentionHead_16260746182865
// MI455X (gfx1250) — matrix operand order repaired and checked by reading, not compiled, not run on hardware
//
#include <hip/hip_runtime.h>

typedef __attribute__((ext_vector_type(16))) _Float16 v16h;
typedef __attribute__((ext_vector_type(8)))  _Float16 v8h;
typedef __attribute__((ext_vector_type(8)))  float    v8f;
typedef __attribute__((ext_vector_type(4)))  float    v4f;

#define B_  4
#define T_  4096
#define D_  1024
#define DA_ 64

union FragH { v16h v; v8h h[2]; _Float16 e[16]; };

static __device__ __forceinline__ v8f wmma_f16(const FragH& a, const FragH& b, v8f c) {
    return __builtin_amdgcn_wmma_f32_16x16x32_f16(false, a.v, false, b.v, (short)0, c, false, false);
}

// ---------------------------------------------------------------------------
// Kernel 0: pre-swizzle W{q,k,v} (fp32 [1024,64]) into f16 B-fragment order:
//   Wf[((p*32 + chunk)*4 + ntile)*512 + lane*16 + h]
//     = (f16) W_p[32*chunk + h + 16*(lane>>4)][16*ntile + (lane&15)]
// so the GEMM loop loads each 32x16 B-fragment as two contiguous b128s.
// ---------------------------------------------------------------------------
__global__ __launch_bounds__(256) void wswizzle_kernel(
    const float* __restrict__ Wq, const float* __restrict__ Wk,
    const float* __restrict__ Wv, _Float16* __restrict__ Wf)
{
    int i = blockIdx.x * 256 + threadIdx.x;      // 0 .. 3*65536-1
    int h = i & 15;
    int l = (i >> 4) & 31;
    int t = (i >> 9) & 3;
    int c = (i >> 11) & 31;
    int p = i >> 16;
    const float* W = (p == 0) ? Wq : (p == 1) ? Wk : Wv;
    int hl = l >> 4, ln = l & 15;
    int k  = 32 * c + h + 16 * hl;
    Wf[i] = (_Float16)W[(size_t)k * DA_ + 16 * t + ln];
}

// ---------------------------------------------------------------------------
// Kernel 1: QKV projection.  One wave computes one 16-row tile of one of
// {Q, K, V}.  Q,K stored f16 row-major [B*T, 64]; V stored f16 transposed
// [B, 64, T] so attention's PV B-fragments are contiguous loads.
// ---------------------------------------------------------------------------
__global__ __launch_bounds__(256) void qkv_proj_kernel(
    const float* __restrict__ x,
    const float* __restrict__ bq, const float* __restrict__ bk,
    const float* __restrict__ bv, const _Float16* __restrict__ Wf,
    _Float16* __restrict__ Qh, _Float16* __restrict__ Kh,
    _Float16* __restrict__ Vt)
{
    const int lane = threadIdx.x & 31;
    const int wave = threadIdx.x >> 5;
    const int ln   = lane & 15;
    const int hl   = lane >> 4;

    const int wid  = blockIdx.x * 8 + wave;          // 0 .. 3071
    const int p    = wid % 3;                        // 0=Q 1=K 2=V
    const int tile = wid / 3;                        // 0 .. 1023
    if (tile >= (B_ * T_) / 16) return;

    const float* bias = (p == 0) ? bq : (p == 1) ? bk : bv;

    v8f acc[4];
    #pragma unroll
    for (int t = 0; t < 4; ++t) acc[t] = (v8f){};

    const float* xrow = x + (size_t)(tile * 16 + ln) * D_;

    for (int d0 = 0; d0 < D_; d0 += 32) {
        // A fragment: x tile 16x32 fp32 -> f16.
        // e[h] <-> K = h + 16*hl  (contiguous, matches pre-swizzled B)
        FragH a;
        const v4f* g0 = (const v4f*)(xrow + d0 + 16 * hl);
        const v4f* g1 = (const v4f*)(xrow + d0 + 16 * hl + 8);
        v4f x0 = g0[0], x1 = g0[1], x2 = g1[0], x3 = g1[1];
        #pragma unroll
        for (int i = 0; i < 4; ++i) {
            a.e[i]      = (_Float16)x0[i];
            a.e[4 + i]  = (_Float16)x1[i];
            a.e[8 + i]  = (_Float16)x2[i];
            a.e[12 + i] = (_Float16)x3[i];
        }
        // Pre-swizzled B fragments: 2 contiguous b128 loads each.
        const _Float16* wchunk =
            Wf + ((size_t)(p * 32 + (d0 >> 5)) * 4) * 512 + lane * 16;
        #pragma unroll
        for (int t = 0; t < 4; ++t) {
            FragH bm;
            bm.h[0] = *(const v8h*)(wchunk + (size_t)t * 512);
            bm.h[1] = *(const v8h*)(wchunk + (size_t)t * 512 + 8);
            acc[t] = wmma_f16(a, bm, acc[t]);
        }
    }

    const int rowbase = tile * 16 + 8 * hl;          // M = r + 8*hl
    #pragma unroll
    for (int t = 0; t < 4; ++t) {
        float bval = bias[16 * t + ln];
        #pragma unroll
        for (int r = 0; r < 8; ++r) {
            float    val = acc[t][r] + bval;
            _Float16 hv  = (_Float16)val;
            int      rg  = rowbase + r;              // flat row in [0, B*T)
            if (p == 2) {
                int bb = rg >> 12;                   // / T_
                int tl = rg & (T_ - 1);
                Vt[(size_t)bb * (DA_ * T_) + (size_t)(16 * t + ln) * T_ + tl] = hv;
            } else {
                _Float16* dst = (p == 0) ? Qh : Kh;
                dst[(size_t)rg * DA_ + 16 * t + ln] = hv;
            }
        }
    }
}

// ---------------------------------------------------------------------------
// Kernel 2: causal flash attention.  8 waves/block own 8 consecutive query
// tiles of one batch.  K/V 32-key tiles staged block-wide into LDS with
// double-buffered global_load_async_to_lds_b128 (ASYNCcnt), so each tile is
// pulled from L2 once per block instead of once per wave.
// ---------------------------------------------------------------------------
__global__ __launch_bounds__(256) void flash_attn_kernel(
    const _Float16* __restrict__ Qh, const _Float16* __restrict__ Kh,
    const _Float16* __restrict__ Vt, float* __restrict__ out)
{
    __shared__ _Float16 Ks[2][32 * 64];   // [buf][key][a]      4 KB each
    __shared__ _Float16 Vs[2][64 * 32];   // [buf][a][key]      4 KB each
    __shared__ _Float16 Pl[8 * 512];      // per-wave P transpose scratch

    const int tid  = threadIdx.x;
    const int lane = tid & 31;
    const int wave = tid >> 5;
    const int ln   = lane & 15;
    const int hl   = lane >> 4;

    const int tile = blockIdx.x * 8 + wave;          // 0 .. 1023
    const int qg0  = tile * 16;                      // flat row of first query
    const int b    = (blockIdx.x * 128) >> 12;       // batch (uniform in block)
    const int ql0  = qg0 & (T_ - 1);                 // local t of first query
    const int kend_w   = ql0 + 16;                   // this wave's key bound
    const int kend_max = ((blockIdx.x * 128) & (T_ - 1)) + 128;  // block bound

    _Float16* pbuf = Pl + wave * 512;

    const _Float16* Kbase = Kh + (size_t)b * T_ * DA_;
    const _Float16* Vbase = Vt + (size_t)b * DA_ * T_;

    // Cooperative async copy: 256 threads x 16B = one 4 KB tile per issue.
    auto issue_copies = [&](int j0, int buf) {
        const _Float16* kg = Kbase + (size_t)(j0 + (tid >> 3)) * DA_ + (tid & 7) * 8;
        unsigned kl = (unsigned)(uintptr_t)(&Ks[buf][0]) + (unsigned)tid * 16u;
        asm volatile("global_load_async_to_lds_b128 %0, %1, off"
                     :: "v"(kl), "v"((unsigned long long)(uintptr_t)kg) : "memory");
        const _Float16* vg = Vbase + (size_t)(tid >> 2) * T_ + j0 + (tid & 3) * 8;
        unsigned vl = (unsigned)(uintptr_t)(&Vs[buf][0]) + (unsigned)tid * 16u;
        asm volatile("global_load_async_to_lds_b128 %0, %1, off"
                     :: "v"(vl), "v"((unsigned long long)(uintptr_t)vg) : "memory");
    };

    // Q A-fragments (two 16x32 chunks over DA=64), persistent in registers.
    FragH q0, q1;
    {
        const _Float16* qrow = Qh + (size_t)(qg0 + ln) * DA_;
        q0.h[0] = *(const v8h*)(qrow + 16 * hl);
        q0.h[1] = *(const v8h*)(qrow + 16 * hl + 8);
        q1.h[0] = *(const v8h*)(qrow + 32 + 16 * hl);
        q1.h[1] = *(const v8h*)(qrow + 32 + 16 * hl + 8);
    }

    v8f o[4];
    #pragma unroll
    for (int t = 0; t < 4; ++t) o[t] = (v8f){};
    float mrow[8], lrow[8];
    #pragma unroll
    for (int r = 0; r < 8; ++r) { mrow[r] = -1e30f; lrow[r] = 0.0f; }

    const float scale = 1.0f / 96.0f;                // 1/(sqrt(64)*12)

    issue_copies(0, 0);
    int cur = 0;
    for (int j0 = 0; j0 < kend_max; j0 += 32, cur ^= 1) {
        // Prefetch next tile into the other buffer, then wait for current.
        if (j0 + 32 < kend_max) {
            issue_copies(j0 + 32, cur ^ 1);
            asm volatile("s_wait_asynccnt 0x2" ::: "memory");
        } else {
            asm volatile("s_wait_asynccnt 0x0" ::: "memory");
        }
        __syncthreads();                              // current tile visible

        if (j0 < kend_w) {
            // ---- S = scale * Q K^T  (16 queries x 32 keys), K from LDS ----
            v8f s[2];
            #pragma unroll
            for (int sub = 0; sub < 2; ++sub) {
                FragH kb0, kb1;
                const _Float16* kr = &Ks[cur][(16 * sub + ln) * 64];
                kb0.h[0] = *(const v8h*)(kr + 16 * hl);
                kb0.h[1] = *(const v8h*)(kr + 16 * hl + 8);
                kb1.h[0] = *(const v8h*)(kr + 32 + 16 * hl);
                kb1.h[1] = *(const v8h*)(kr + 32 + 16 * hl + 8);
                v8f a = (v8f){};
                a = wmma_f16(q0, kb0, a);
                a = wmma_f16(q1, kb1, a);
                s[sub] = a;
            }
            #pragma unroll
            for (int c = 0; c < 2; ++c)
                #pragma unroll
                for (int r = 0; r < 8; ++r) s[c][r] *= scale;

            // ---- causal mask (diagonal blocks only) ----
            if (j0 + 31 > ql0) {
                #pragma unroll
                for (int c = 0; c < 2; ++c) {
                    int k = j0 + ln + 16 * c;
                    #pragma unroll
                    for (int r = 0; r < 8; ++r) {
                        int q = ql0 + r + 8 * hl;
                        if (k > q) s[c][r] = -1e30f;
                    }
                }
            }

            // ---- online softmax (row reductions over 16-lane halves) ----
            float p0[8], p1[8], alpha[8];
            #pragma unroll
            for (int r = 0; r < 8; ++r) {
                float v = fmaxf(s[0][r], s[1][r]);
                v = fmaxf(v, __shfl_xor(v, 8, 32));
                v = fmaxf(v, __shfl_xor(v, 4, 32));
                v = fmaxf(v, __shfl_xor(v, 2, 32));
                v = fmaxf(v, __shfl_xor(v, 1, 32));
                float mn = fmaxf(mrow[r], v);
                alpha[r] = __expf(mrow[r] - mn);
                mrow[r]  = mn;
                p0[r] = __expf(s[0][r] - mn);
                p1[r] = __expf(s[1][r] - mn);
                float rs = p0[r] + p1[r];
                rs += __shfl_xor(rs, 8, 32);
                rs += __shfl_xor(rs, 4, 32);
                rs += __shfl_xor(rs, 2, 32);
                rs += __shfl_xor(rs, 1, 32);
                lrow[r] = lrow[r] * alpha[r] + rs;
            }
            #pragma unroll
            for (int t = 0; t < 4; ++t)
                #pragma unroll
                for (int r = 0; r < 8; ++r) o[t][r] *= alpha[r];

            // ---- C-layout -> A-layout transpose of P via per-wave LDS ----
            #pragma unroll
            for (int r = 0; r < 8; ++r) {
                int m = r + 8 * hl;
                pbuf[m * 32 + ln]      = (_Float16)p0[r];
                pbuf[m * 32 + 16 + ln] = (_Float16)p1[r];
            }
            asm volatile("s_wait_dscnt 0" ::: "memory");
            FragH pa;
            pa.h[0] = *(const v8h*)(pbuf + ln * 32 + 16 * hl);
            pa.h[1] = *(const v8h*)(pbuf + ln * 32 + 16 * hl + 8);

            // ---- O += P * V, V fragments from LDS ----
            #pragma unroll
            for (int t = 0; t < 4; ++t) {
                FragH vb;
                const _Float16* vr = &Vs[cur][(16 * t + ln) * 32 + 16 * hl];
                vb.h[0] = *(const v8h*)(vr);
                vb.h[1] = *(const v8h*)(vr + 8);
                o[t] = wmma_f16(pa, vb, o[t]);
            }
        }
        __syncthreads();                              // protect buf before reuse
    }

    // ---- normalize and store fp32 output [B, T, 64] ----
    #pragma unroll
    for (int t = 0; t < 4; ++t)
        #pragma unroll
        for (int r = 0; r < 8; ++r) {
            int rg = qg0 + r + 8 * hl;
            out[(size_t)rg * DA_ + 16 * t + ln] = o[t][r] / lrow[r];
        }
}

extern "C" void kernel_launch(void* const* d_in, const int* in_sizes, int n_in,
                              void* d_out, int out_size, void* d_ws, size_t ws_size,
                              hipStream_t stream) {
    const float* x  = (const float*)d_in[0];
    const float* Wq = (const float*)d_in[1];
    const float* bq = (const float*)d_in[2];
    const float* Wk = (const float*)d_in[3];
    const float* bk = (const float*)d_in[4];
    const float* Wv = (const float*)d_in[5];
    const float* bv = (const float*)d_in[6];
    float* out = (float*)d_out;

    // Workspace: Q,K f16 [B*T,64] + V f16 transposed [B,64,T] + swizzled W.
    _Float16* Qh = (_Float16*)d_ws;
    _Float16* Kh = Qh + (size_t)B_ * T_ * DA_;
    _Float16* Vt = Kh + (size_t)B_ * T_ * DA_;
    _Float16* Wf = Vt + (size_t)B_ * DA_ * T_;       // 3*1024*64 halfs

    wswizzle_kernel<<<(3 * D_ * DA_) / 256, 256, 0, stream>>>(Wq, Wk, Wv, Wf);

    const int proj_blocks = (3 * (B_ * T_ / 16)) / 8;      // 384
    qkv_proj_kernel<<<proj_blocks, 256, 0, stream>>>(x, bq, bk, bv, Wf,
                                                     Qh, Kh, Vt);

    const int attn_blocks = (B_ * T_ / 16) / 8;            // 128
    flash_attn_kernel<<<attn_blocks, 256, 0, stream>>>(Qh, Kh, Vt, out);
}
